// Model_13907104104903
// MI455X (gfx1250) — matrix operand order repaired and checked by reading, not compiled, not run on hardware
//
#include <hip/hip_runtime.h>
#include <hip/hip_bf16.h>

typedef __attribute__((ext_vector_type(16))) _Float16 v16h;
typedef __attribute__((ext_vector_type(8)))  _Float16 v8h;
typedef __attribute__((ext_vector_type(4)))  _Float16 v4h;
typedef __attribute__((ext_vector_type(8)))  float    v8f;
typedef __attribute__((ext_vector_type(4)))  float    v4f;

#define BB 4
#define SS 256
#define DD 768
#define VV 50257
#define ST_ 64
#define EN_ 255
#define LL 192          // S - ST
#define PGR 191         // EN - ST rows with p_gen

// ---------------- f32 -> f16 convert (vectorized, n % 4 == 0) ----------------
__global__ void cvt_f16_kernel(const float* __restrict__ src, _Float16* __restrict__ dst, long n4) {
    long i = (long)blockIdx.x * blockDim.x + threadIdx.x;
    long stride = (long)gridDim.x * blockDim.x;
    const v4f* s4 = (const v4f*)src;
    v4h* d4 = (v4h*)dst;
    for (; i < n4; i += stride)
        d4[i] = __builtin_convertvector(s4[i], v4h);
}

// Stage one 32x16-half B tile (1KB) into LDS via the gfx1250 async-to-LDS path.
// chunk c (0..63) = 16 bytes: row = c>>1 (K), half = c&1 (N cols 0..7 / 8..15).
// Leading s_wait_dscnt 0 guarantees prior ds_load gathers of the destination
// buffer have retired before the async writes can land (WAR on LDS).
__device__ __forceinline__ void stage_btile_async(const _Float16* Wb, int ldw, int kb, int n0,
                                                  unsigned ldsdst, int lane) {
    int c0 = lane, c1 = lane + 32;
    const _Float16* g0 = Wb + (long)(kb + (c0 >> 1)) * ldw + n0 + (c0 & 1) * 8;
    const _Float16* g1 = Wb + (long)(kb + (c1 >> 1)) * ldw + n0 + (c1 & 1) * 8;
    unsigned l0 = ldsdst + (unsigned)c0 * 16u;
    unsigned l1 = ldsdst + (unsigned)c1 * 16u;
    asm volatile("s_wait_dscnt 0\n\t"
                 "global_load_async_to_lds_b128 %0, %1, off\n\t"
                 "global_load_async_to_lds_b128 %2, %3, off"
                 :: "v"(l0), "v"((unsigned long long)(size_t)g0),
                    "v"(l1), "v"((unsigned long long)(size_t)g1)
                 : "memory");
}

// ---------------- WMMA f16 GEMM: out[M,N] = A[M,K] * W[K,N] (+bias) ----------------
// 128 threads = 4 waves; wave w owns a 64x16 output strip (4 M-tiles, ntile = blockIdx.x*4+w)
// with two private 1KB LDS slices for double-buffered async B staging.
// Per K-step: 1 B transpose-gather feeds 4 WMMAs (A fragments are contiguous b128 loads).
// Fragment layouts per CDNA5 ISA 7.12.2 (wave32):
//   A 16x32: lane<16 row=lane holds K = {0..7,16..23}; lane>=16 row=lane-16 holds K={8..15,24..31}
//   B 32x16: lane holds col=lane&15; halves i=0..15 -> K = (lane>>4)*8 + (i&7) + (i>>3)*16  (LDS gather)
//   C/D f32: vgpr r, lane -> (M = r + 8*(lane>>4), N = lane&15)
__global__ void gemm_wmma_f16(const _Float16* __restrict__ A, int lda, long strideA,
                              const _Float16* __restrict__ W, int ldw, long strideW,
                              const float* __restrict__ bias,
                              float* __restrict__ out, int ldo, long strideOut,
                              int K) {
    __shared__ __align__(1024) _Float16 btile[4][2][32 * 16];   // 4 waves x 2 buffers x 1KB

    int b = blockIdx.z;
    const _Float16* Ab = A + (long)b * strideA;
    const _Float16* Wb = W + (long)b * strideW;
    float* Ob = out + (long)b * strideOut;

    int lane = threadIdx.x & 31;
    int w    = threadIdx.x >> 5;
    int ln15 = lane & 15;
    int hi   = lane >> 4;
    int m0 = blockIdx.y << 6;                     // 64-row M strip
    int n0 = (blockIdx.x * 4 + w) << 4;

    _Float16* t0 = &btile[w][0][0];
    _Float16* t1 = &btile[w][1][0];
    unsigned lds0 = (unsigned)(size_t)t0;
    unsigned lds1 = (unsigned)(size_t)t1;

    float bv = bias ? bias[n0 + ln15] : 0.0f;
    v8f acc[4];
#pragma unroll
    for (int j = 0; j < 4; ++j)
#pragma unroll
        for (int r = 0; r < 8; ++r) acc[j][r] = bv;

    const int nk = K >> 5;
    stage_btile_async(Wb, ldw, 0, n0, lds0, lane);

    for (int kbi = 0; kbi < nk; ++kbi) {
        int kb = kbi << 5;
        const _Float16* tcur = (kbi & 1) ? t1 : t0;
        unsigned ldsnxt      = (kbi & 1) ? lds0 : lds1;

        if (kbi + 1 < nk) {
            stage_btile_async(Wb, ldw, kb + 32, n0, ldsnxt, lane);
            asm volatile("s_wait_asynccnt 2" ::: "memory");   // oldest buffer landed
        } else {
            asm volatile("s_wait_asynccnt 0" ::: "memory");
        }

        // ---- B fragment: K-strided transpose gather from LDS (reused by 4 WMMAs) ----
        v16h bf;
#pragma unroll
        for (int i = 0; i < 16; ++i)
            bf[i] = tcur[(hi * 8 + (i & 7) + (i >> 3) * 16) * 16 + ln15];

        // ---- 4 A fragments (contiguous b128 loads) + 4 WMMAs ----
#pragma unroll
        for (int j = 0; j < 4; ++j) {
            const _Float16* arow = Ab + (long)(m0 + j * 16 + ln15) * lda;
            v8h alo = *(const v8h*)(arow + kb + hi * 8);        // K = hi*8 .. +7
            v8h ahi = *(const v8h*)(arow + kb + hi * 8 + 16);   // K = hi*8+16 .. +23
            v16h af = __builtin_shufflevector(alo, ahi,
                                              0, 1, 2, 3, 4, 5, 6, 7,
                                              8, 9, 10, 11, 12, 13, 14, 15);
            acc[j] = __builtin_amdgcn_wmma_f32_16x16x32_f16(false, af, false, bf,
                                                            (short)0, acc[j], false, false);
        }
    }
#pragma unroll
    for (int j = 0; j < 4; ++j)
#pragma unroll
        for (int r = 0; r < 8; ++r)
            Ob[(long)(m0 + j * 16 + r + 8 * hi) * ldo + n0 + ln15] = acc[j][r];
}

// ---------------- attention scores + softmax over 64 keys ----------------
__global__ void attn_kernel(const float* __restrict__ q, const float* __restrict__ kmat,
                            const float* __restrict__ Va_w, const float* __restrict__ Va_b,
                            float* __restrict__ wts, _Float16* __restrict__ wtsH) {
    int l = blockIdx.x, b = blockIdx.y;
    int tid = threadIdx.x, lane = tid & 31, w = tid >> 5;
    __shared__ float sc[64];
    __shared__ float sw[64];

    const float* qr = q + ((long)b * LL + l) * DD;
    const float* kb = kmat + (long)b * ST_ * DD;

    for (int kk = w * 8; kk < w * 8 + 8; ++kk) {
        const float* kr = kb + (long)kk * DD;
        float p = 0.0f;
        for (int d = lane; d < DD; d += 32)
            p += tanhf(qr[d] + kr[d]) * Va_w[d];
#pragma unroll
        for (int off = 16; off; off >>= 1) p += __shfl_xor(p, off, 32);
        if (lane == 0) sc[kk] = p + Va_b[0];
    }
    __syncthreads();
    if (tid == 0) {
        float m = sc[0];
        for (int i = 1; i < 64; ++i) m = fmaxf(m, sc[i]);
        float s = 0.0f;
        for (int i = 0; i < 64; ++i) { float e = __expf(sc[i] - m); sw[i] = e; s += e; }
        float inv = 1.0f / s;
        for (int i = 0; i < 64; ++i) sw[i] *= inv;
    }
    __syncthreads();
    if (tid < 64) {
        long o = ((long)b * LL + l) * 64 + tid;
        wts[o]  = sw[tid];
        wtsH[o] = (_Float16)sw[tid];
    }
}

// ---------------- p_gen for positions r = ST..EN-1 ----------------
__global__ void pgen_kernel(const float* __restrict__ ctx, const float* __restrict__ hidden,
                            const float* __restrict__ wte, const int* __restrict__ toks,
                            const float* __restrict__ Wh_w, const float* __restrict__ Wh_b,
                            const float* __restrict__ Ws_w, const float* __restrict__ Ws_b,
                            const float* __restrict__ Wx_w, const float* __restrict__ Wx_b,
                            float* __restrict__ pgen) {
    int r = ST_ + blockIdx.x;          // 64..254
    int b = blockIdx.y;
    int tid = threadIdx.x;
    const float* hrow = hidden + ((long)b * SS + r) * DD;
    int tok = toks[b * SS + r];
    const float* erow = wte + (long)tok * DD;
    const float* crow = (r < LL) ? (ctx + ((long)b * LL + r) * DD) : nullptr;

    float part = 0.0f;
    for (int d = tid; d < DD; d += 256) {
        float a = hrow[d] * Ws_w[d] + erow[d] * Wx_w[d];
        if (crow) a += crow[d] * Wh_w[d];
        part += a;
    }
#pragma unroll
    for (int off = 16; off; off >>= 1) part += __shfl_xor(part, off, 32);
    __shared__ float red[8];
    if ((tid & 31) == 0) red[tid >> 5] = part;
    __syncthreads();
    if (tid == 0) {
        float s = 0.0f;
        for (int i = 0; i < 8; ++i) s += red[i];
        s += Wh_b[0] + Ws_b[0] + Wx_b[0];
        pgen[b * PGR + blockIdx.x] = 1.0f / (1.0f + __expf(-s));
    }
}

// ---------------- last-occurrence mask over first 64 tokens ----------------
__global__ void lastocc_kernel(const int* __restrict__ toks, int* __restrict__ lo) {
    int b = blockIdx.x, k = threadIdx.x;
    int t = toks[b * SS + k];
    int m = 1;
    for (int j = k + 1; j < ST_; ++j)
        if (toks[b * SS + j] == t) m = 0;
    lo[b * ST_ + k] = m;
}

// ---------------- fused row kernel: softmax(logits) blend + pointer scatter ----------------
__global__ void row_kernel(const float* __restrict__ logits, const float* __restrict__ pgen,
                           const float* __restrict__ wts, const int* __restrict__ lastocc,
                           const int* __restrict__ toks, float* __restrict__ out) {
    int s = blockIdx.x, b = blockIdx.y, tid = threadIdx.x;
    float* orow = out + ((long)b * SS + s) * (long)VV;
    if (s < ST_ || s == EN_) {           // zero rows (row 255 is all-zero per reference)
        for (long v = tid; v < VV; v += 256) orow[v] = 0.0f;
        return;
    }
    const float* lrow = logits + ((long)b * SS + s) * (long)VV;

    // online softmax statistics
    float m = -3.4e38f, sum = 0.0f;
    for (long v = tid; v < VV; v += 256) {
        if (v + 8192 < VV) __builtin_prefetch(lrow + v + 8192, 0, 0);
        float x = lrow[v];
        float nm = fmaxf(m, x);
        sum = sum * __expf(m - nm) + __expf(x - nm);
        m = nm;
    }
    __shared__ float sm[256], ss[256];
    sm[tid] = m; ss[tid] = sum;
    __syncthreads();
    if (tid == 0) {
        float M = sm[0], S = ss[0];
        for (int i = 1; i < 256; ++i) {
            float nm = fmaxf(M, sm[i]);
            S = S * __expf(M - nm) + ss[i] * __expf(sm[i] - nm);
            M = nm;
        }
        sm[0] = M; ss[0] = S;
    }
    __syncthreads();
    float M = sm[0], invS = 1.0f / ss[0];
    float pg = pgen[b * PGR + (s - ST_)];

    for (long v = tid; v < VV; v += 256)
        orow[v] = pg * __expf(lrow[v] - M) * invS;

    __syncthreads();
    // pointer scatter: only positions s < 192 have nonzero attn rows (ref places row l at pos l)
    if (s < LL && tid < ST_ && lastocc[b * ST_ + tid]) {
        int vtok = toks[b * SS + tid];
        orow[vtok] += (1.0f - pg) * wts[((long)b * LL + s) * 64 + tid];
    }
}

extern "C" void kernel_launch(void* const* d_in, const int* in_sizes, int n_in,
                              void* d_out, int out_size, void* d_ws, size_t ws_size,
                              hipStream_t stream) {
    const float* hidden = (const float*)d_in[0];
    const float* logits = (const float*)d_in[1];
    const float* wte    = (const float*)d_in[2];
    const float* Wa_w   = (const float*)d_in[3];
    const float* Wa_b   = (const float*)d_in[4];
    const float* Ua_w   = (const float*)d_in[5];
    const float* Ua_b   = (const float*)d_in[6];
    const float* Va_w   = (const float*)d_in[7];
    const float* Va_b   = (const float*)d_in[8];
    const float* Wh_w   = (const float*)d_in[9];
    const float* Wh_b   = (const float*)d_in[10];
    const float* Ws_w   = (const float*)d_in[11];
    const float* Ws_b   = (const float*)d_in[12];
    const float* Wx_w   = (const float*)d_in[13];
    const float* Wx_b   = (const float*)d_in[14];
    const int*   toks   = (const int*)d_in[15];
    float* out = (float*)d_out;

    char* base = (char*)d_ws;
    size_t off = 0;
    auto alloc = [&](size_t bytes) { size_t r = off; off = (off + bytes + 255) & ~(size_t)255; return r; };
    _Float16* hidH = (_Float16*)(base + alloc((size_t)BB * SS * DD * 2));
    _Float16* WaH  = (_Float16*)(base + alloc((size_t)DD * DD * 2));
    _Float16* UaH  = (_Float16*)(base + alloc((size_t)DD * DD * 2));
    float*    qb   = (float*)   (base + alloc((size_t)BB * LL * DD * 4));
    float*    kb   = (float*)   (base + alloc((size_t)BB * ST_ * DD * 4));
    float*    wts  = (float*)   (base + alloc((size_t)BB * LL * 64 * 4));
    _Float16* wtsH = (_Float16*)(base + alloc((size_t)BB * LL * 64 * 2));
    float*    ctx  = (float*)   (base + alloc((size_t)BB * LL * DD * 4));
    float*    pgen = (float*)   (base + alloc((size_t)BB * PGR * 4));
    int*      lo   = (int*)     (base + alloc((size_t)BB * ST_ * 4));

    // 1) convert activations + weights to f16 (vectorized)
    cvt_f16_kernel<<<512, 256, 0, stream>>>(hidden, hidH, (long)BB * SS * DD / 4);
    cvt_f16_kernel<<<512, 256, 0, stream>>>(Wa_w, WaH, (long)DD * DD / 4);
    cvt_f16_kernel<<<512, 256, 0, stream>>>(Ua_w, UaH, (long)DD * DD / 4);

    // 2) q = summary @ Wa + Wa_b   (M=192, K=768, N=768): grid (N/64, M/64, B)
    gemm_wmma_f16<<<dim3(DD / 64, LL / 64, BB), 128, 0, stream>>>(
        hidH + (long)ST_ * DD, DD, (long)SS * DD,
        WaH, DD, 0, Wa_b, qb, DD, (long)LL * DD, DD);

    // 3) k = source @ Ua + Ua_b    (M=64, K=768, N=768)
    gemm_wmma_f16<<<dim3(DD / 64, ST_ / 64, BB), 128, 0, stream>>>(
        hidH, DD, (long)SS * DD,
        UaH, DD, 0, Ua_b, kb, DD, (long)ST_ * DD, DD);

    // 4) additive-attention scores + softmax -> weights (f32 + f16)
    attn_kernel<<<dim3(LL, BB), 256, 0, stream>>>(qb, kb, Va_w, Va_b, wts, wtsH);

    // 5) ctx = weights @ source    (M=192, K=64, N=768) via WMMA
    gemm_wmma_f16<<<dim3(DD / 64, LL / 64, BB), 128, 0, stream>>>(
        wtsH, 64, (long)LL * 64,
        hidH, DD, (long)SS * DD,
        nullptr, ctx, DD, (long)LL * DD, 64);

    // 6) p_gen for positions 64..254
    pgen_kernel<<<dim3(PGR, BB), 256, 0, stream>>>(ctx, hidden, wte, toks,
                                                   Wh_w, Wh_b, Ws_w, Ws_b, Wx_w, Wx_b, pgen);

    // 7) last-occurrence mask
    lastocc_kernel<<<BB, ST_, 0, stream>>>(toks, lo);

    // 8) fused output rows (zero / softmax blend / pointer scatter)
    row_kernel<<<dim3(SS, BB), 256, 0, stream>>>(logits, pgen, wts, lo, toks, out);
}
